// Model_27479200759938
// MI455X (gfx1250) — matrix operand order repaired and checked by reading, not compiled, not run on hardware
//
#include <hip/hip_runtime.h>
#include <math.h>

// Problem constants (match reference)
#define NC      8192
#define SDIM    32
#define MDIM    8
#define HDIM    64
#define FOUR_PI 12.566370614359172f

typedef __attribute__((ext_vector_type(16))) _Float16 v16h;
typedef __attribute__((ext_vector_type(8)))  float    v8f;

// ---------------------------------------------------------------------------
// Wave-level 16x16 output-tile GEMM using V_WMMA_F32_16X16X32_F16.
//   A  : f16 LDS, row-major, stride lda; rows [row0, row0+16)
//   Wh : f16 LDS, 64x64 zero-padded weight (out_features x in_features), so
//        B[k][n] = Wh[n0+n][k]  (computes X @ W^T like the reference Linear)
//   Y  : f32 LDS, stride ldy; 16x16 tile written at (row0, n0)
// A-layout (16-bit A 16x32): lane m(+16) holds K = {0..7,16..23} (+8 hi-half)
// B-layout (16-bit B 32x16): lane n(+16) holds K = {0..7,16..23} (+8 hi-half)
// C-layout: VGPR r, lane n -> (row r + 8*hi, col n)
// ---------------------------------------------------------------------------
__device__ __forceinline__
void wmma_tile(const _Float16* A, int lda, int row0,
               const _Float16* Wh, int n0,
               float* Y, int ldy, int ktiles, int lane)
{
    const int m  = lane & 15;
    const int hi = lane >> 4;
    v8f c = {};
    for (int kt = 0; kt < ktiles; ++kt) {
        const int kb = kt * 32;
        v16h a, b;
#pragma unroll
        for (int e = 0; e < 16; ++e) {
            const int ka = kb + ((e < 8) ? e : e + 8) + hi * 8;
            a[e] = A[(row0 + m) * lda + ka];
            const int kb2 = kb + ((e < 8) ? e : e + 8) + hi * 8;
            b[e] = Wh[(n0 + m) * 64 + kb2];
        }
#if defined(__gfx1250__)
        c = __builtin_amdgcn_wmma_f32_16x16x32_f16(
                /*neg_a=*/false, a, /*neg_b=*/false, b,
                /*c_mod=*/(short)0, c, /*reuse_a=*/false, /*reuse_b=*/false);
#endif
    }
#pragma unroll
    for (int r = 0; r < 8; ++r)
        Y[(row0 + r + hi * 8) * ldy + (n0 + m)] = c[r];
}

// ---------------------------------------------------------------------------
__global__ void k_zero(float* __restrict__ p, int n)
{
    int i = blockIdx.x * blockDim.x + threadIdx.x;
    if (i < n) p[i] = 0.f;
}

// ---------------------------------------------------------------------------
// Secretion MLP: sec' = relu(MLP_LN(state)) * act * 1/(4*pi*D_m)
// Also precompute per-cell exp(radius) and 1/radius for the pairwise kernel.
// Block = 128 threads (4 waves) handles 64 cells.
// ---------------------------------------------------------------------------
__global__ __launch_bounds__(128)
void k_secretion(const float* __restrict__ state, const float* __restrict__ radius,
                 const unsigned char* __restrict__ active,
                 const float* __restrict__ dif,
                 const float* __restrict__ W1, const float* __restrict__ b1,
                 const float* __restrict__ g,  const float* __restrict__ h,
                 const float* __restrict__ W2, const float* __restrict__ b2,
                 float* __restrict__ sec_scaled, float* __restrict__ Ebuf,
                 float* __restrict__ invrad)
{
    __shared__ _Float16 Xh[64 * 64];
    __shared__ float    Yf[64 * 64];
    __shared__ _Float16 Yh[64 * 64];
    __shared__ float    Zf[64 * 16];
    __shared__ _Float16 Wh[64 * 64];

    const int tid  = threadIdx.x;
    const int lane = tid & 31;
    const int wv   = tid >> 5;
    const int c0   = blockIdx.x * 64;

    for (int idx = tid; idx < 64 * 64; idx += 128) {
        const int r = idx >> 6, k = idx & 63;
        Xh[idx] = (_Float16)((k < SDIM) ? state[(c0 + r) * SDIM + k] : 0.f);
        Wh[idx] = (_Float16)((k < SDIM) ? W1[r * SDIM + k] : 0.f);
    }
    __syncthreads();
    for (int nt = 0; nt < 4; ++nt)
        wmma_tile(Xh, 64, wv * 16, Wh, nt * 16, Yf, 64, 1, lane);
    __syncthreads();
    // bias + layernorm + relu -> Yh; restage W2 (8x64, zero-padded rows)
    for (int r = tid; r < 64; r += 128) {
        float mu = 0.f, s2 = 0.f;
        for (int q = 0; q < HDIM; ++q) {
            float v = Yf[r * 64 + q] + b1[q];
            Yf[r * 64 + q] = v; mu += v; s2 += v * v;
        }
        mu *= (1.f / HDIM);
        const float inv = rsqrtf(s2 * (1.f / HDIM) - mu * mu + 1e-5f);
        for (int q = 0; q < HDIM; ++q) {
            const float t = g[q] * (Yf[r * 64 + q] - mu) * inv + h[q];
            Yh[r * 64 + q] = (_Float16)fmaxf(t, 0.f);
        }
    }
    for (int idx = tid; idx < 64 * 64; idx += 128) {
        const int r = idx >> 6, k = idx & 63;
        Wh[idx] = (_Float16)((r < MDIM) ? W2[r * 64 + k] : 0.f);
    }
    __syncthreads();
    wmma_tile(Yh, 64, wv * 16, Wh, 0, Zf, 16, 2, lane);
    __syncthreads();
    for (int idx = tid; idx < 64 * MDIM; idx += 128) {
        const int c = idx >> 3, mm = idx & 7;
        const float Dm = fmaxf(dif[mm], 1e-3f);
        const float cm = 1.f / (FOUR_PI * Dm);
        const float am = active[c0 + c] ? 1.f : 0.f;
        const float v  = fmaxf(Zf[c * 16 + mm] + b2[mm], 0.f);
        sec_scaled[(c0 + c) * MDIM + mm] = v * am * cm;
    }
    if (tid < 64) {
        const float r = radius[c0 + tid];
        Ebuf[c0 + tid]   = __expf(r);      // exp(r_i): Morse exp factorization
        invrad[c0 + tid] = 1.f / r;
    }
}

// ---------------------------------------------------------------------------
// Pairwise kernel: concentration accumulation + Morse force accumulators.
// Grid: (NC/256 i-blocks) x (16 j-splits); 1 i-cell per thread; j staged in
// LDS tiles of 128 (broadcast reads, no bank conflicts). Partial sums combined
// across j-splits with float atomics.
// ---------------------------------------------------------------------------
__global__ __launch_bounds__(256)
void k_pairwise(const float* __restrict__ pos, const float* __restrict__ radius,
                const float* __restrict__ dif, const float* __restrict__ deg,
                const unsigned char* __restrict__ active,
                const float* __restrict__ sec_scaled,
                const float* __restrict__ Ebuf, const float* __restrict__ invr,
                float* __restrict__ conc, float* __restrict__ coefsum,
                float* __restrict__ coefp)
{
    __shared__ float sx[128], sy[128], sz[128];
    __shared__ float srad[128], sinv[128], sE[128], sa[128];
    __shared__ float ssec[128][MDIM];

    const int tid = threadIdx.x;
    const int i   = blockIdx.x * blockDim.x + tid;
    const int jlen = NC / gridDim.y;
    const int j0   = blockIdx.y * jlen;

    float invlam[MDIM];
#pragma unroll
    for (int m = 0; m < MDIM; ++m) {
        const float Dm = fmaxf(dif[m], 1e-3f);
        const float km = fmaxf(deg[m], 1e-3f);
        invlam[m] = sqrtf(km / Dm);            // 1/lambda_m
    }
    const float px = pos[i * 3 + 0], py = pos[i * 3 + 1], pz = pos[i * 3 + 2];
    const float Ei = Ebuf[i];

    float acc[MDIM];
#pragma unroll
    for (int m = 0; m < MDIM; ++m) acc[m] = 0.f;
    float cs = 0.f, cpx = 0.f, cpy = 0.f, cpz = 0.f;

    for (int t = 0; t < jlen; t += 128) {
        if (tid < 128) {
            const int j = j0 + t + tid;
            sx[tid] = pos[j * 3 + 0]; sy[tid] = pos[j * 3 + 1]; sz[tid] = pos[j * 3 + 2];
            srad[tid] = radius[j]; sinv[tid] = invr[j]; sE[tid] = Ebuf[j];
            sa[tid] = active[j] ? 1.f : 0.f;
#pragma unroll
            for (int m = 0; m < MDIM; ++m) ssec[tid][m] = sec_scaled[j * MDIM + m];
        }
        __syncthreads();
#pragma unroll 2
        for (int jj = 0; jj < 128; ++jj) {
            const int j = j0 + t + jj;
            const float dx = px - sx[jj], dy = py - sy[jj], dz = pz - sz[jj];
            float d2 = fmaf(dx, dx, fmaf(dy, dy, dz * dz));
            d2 = fmaxf(d2, 1e-12f);
            const float rd = rsqrtf(d2);       // 1/d
            const float d  = d2 * rd;          // d
            const float rj = srad[jj];
            const bool  far = d >= rj;
            const float deff  = far ? d  : rj;
            const float invde = far ? rd : sinv[jj];
            // concentration: K = exp(-deff/lam_m) / (4 pi D_m deff); c_m, act_j
            // already folded into ssec.
#pragma unroll
            for (int m = 0; m < MDIM; ++m)
                acc[m] = fmaf(__expf(-deff * invlam[m]) * invde, ssec[jj][m], acc[m]);
            // Morse: e = exp(-(d - ri - rj)) = exp(ri)*exp(rj)*exp(-d)
            if (i != j) {
                const float e    = __expf(-d) * Ei * sE[jj];
                const float fm   = 2.0f * (e * e - e);      // 2*a*depth*(e^2-e)
                const float coef = fm * rd * sa[jj];        // fmag*mask/d (act_j)
                cs  += coef;
                cpx = fmaf(coef, sx[jj], cpx);
                cpy = fmaf(coef, sy[jj], cpy);
                cpz = fmaf(coef, sz[jj], cpz);
            }
        }
        __syncthreads();
    }
#pragma unroll
    for (int m = 0; m < MDIM; ++m) atomicAdd(&conc[i * MDIM + m], acc[m]);
    atomicAdd(&coefsum[i], cs);
    atomicAdd(&coefp[i * 3 + 0], cpx);
    atomicAdd(&coefp[i * 3 + 1], cpy);
    atomicAdd(&coefp[i * 3 + 2], cpz);
}

// ---------------------------------------------------------------------------
// Sensation -> React -> Motility MLP chain (WMMA), mechanics assembly,
// Euler step, final (N,43) output. Block = 128 threads handles 64 cells.
// ---------------------------------------------------------------------------
__global__ __launch_bounds__(128)
void k_update(const float* __restrict__ pos, const float* __restrict__ state,
              const unsigned char* __restrict__ active,
              const float* __restrict__ Wsen1, const float* __restrict__ bsen1,
              const float* __restrict__ gsen,  const float* __restrict__ hsen,
              const float* __restrict__ Wsen2, const float* __restrict__ bsen2,
              const float* __restrict__ Wr1, const float* __restrict__ br1,
              const float* __restrict__ gr,  const float* __restrict__ hr,
              const float* __restrict__ Wr2, const float* __restrict__ br2,
              const float* __restrict__ Wm1, const float* __restrict__ bm1,
              const float* __restrict__ gm,  const float* __restrict__ hm,
              const float* __restrict__ Wm2, const float* __restrict__ bm2,
              const float* __restrict__ conc, const float* __restrict__ coefsum,
              const float* __restrict__ coefp, float* __restrict__ out)
{
    __shared__ _Float16 Xh[64 * 64];
    __shared__ float    Yf[64 * 64];
    __shared__ _Float16 Yh[64 * 64];
    __shared__ float    Zf[64 * 32];
    __shared__ _Float16 Wh[64 * 64];
    __shared__ float    Sf[64 * 32];   // running f32 state
    __shared__ float    Cc[64 * 8];    // act-masked concentration

    const int tid = threadIdx.x, lane = tid & 31, wv = tid >> 5;
    const int c0 = blockIdx.x * 64;

    auto stageW = [&](const float* W, int rows, int cols) {
        for (int idx = tid; idx < 64 * 64; idx += 128) {
            const int r = idx >> 6, k = idx & 63;
            Wh[idx] = (_Float16)((r < rows && k < cols) ? W[r * cols + k] : 0.f);
        }
    };
    auto lnrelu = [&](const float* b, const float* gg, const float* hh) {
        for (int r = tid; r < 64; r += 128) {
            float mu = 0.f, s2 = 0.f;
            for (int q = 0; q < HDIM; ++q) {
                float v = Yf[r * 64 + q] + b[q];
                Yf[r * 64 + q] = v; mu += v; s2 += v * v;
            }
            mu *= (1.f / HDIM);
            const float inv = rsqrtf(s2 * (1.f / HDIM) - mu * mu + 1e-5f);
            for (int q = 0; q < HDIM; ++q) {
                const float t = gg[q] * (Yf[r * 64 + q] - mu) * inv + hh[q];
                Yh[r * 64 + q] = (_Float16)fmaxf(t, 0.f);
            }
        }
    };
    auto gemm1 = [&](int ktiles) {   // Xh x Wh^T -> Yf (64x64)
        for (int nt = 0; nt < 4; ++nt)
            wmma_tile(Xh, 64, wv * 16, Wh, nt * 16, Yf, 64, ktiles, lane);
    };
    auto gemm2 = [&](int ntiles) {   // Yh x Wh^T -> Zf (64x32)
        for (int nt = 0; nt < ntiles; ++nt)
            wmma_tile(Yh, 64, wv * 16, Wh, nt * 16, Zf, 32, 2, lane);
    };
    auto loadX = [&](bool withConc) {
        for (int idx = tid; idx < 64 * 64; idx += 128) {
            const int r = idx >> 6, k = idx & 63;
            float v = 0.f;
            if (k < 32)                   v = Sf[r * 32 + k];
            else if (withConc && k < 40)  v = Cc[r * 8 + (k - 32)];
            Xh[idx] = (_Float16)v;
        }
    };

    for (int idx = tid; idx < 64 * 32; idx += 128)
        Sf[idx] = state[c0 * 32 + idx];
    for (int idx = tid; idx < 64 * 8; idx += 128) {
        const int r = idx >> 3;
        Cc[idx] = conc[c0 * 8 + idx] * (active[c0 + r] ? 1.f : 0.f);
    }
    __syncthreads();

    // ---- sensation: state += MLP([state, conc]) ----
    loadX(true); stageW(Wsen1, HDIM, SDIM + MDIM);
    __syncthreads();
    gemm1(2);
    __syncthreads();
    lnrelu(bsen1, gsen, hsen); stageW(Wsen2, SDIM, HDIM);
    __syncthreads();
    gemm2(2);
    __syncthreads();
    for (int idx = tid; idx < 64 * 32; idx += 128)
        Sf[idx] += Zf[idx] + bsen2[idx & 31];
    __syncthreads();

    // ---- react: state += sigmoid(MLP(state)) ----
    loadX(false); stageW(Wr1, HDIM, SDIM);
    __syncthreads();
    gemm1(1);
    __syncthreads();
    lnrelu(br1, gr, hr); stageW(Wr2, SDIM, HDIM);
    __syncthreads();
    gemm2(2);
    __syncthreads();
    for (int idx = tid; idx < 64 * 32; idx += 128) {
        const float z = Zf[idx] + br2[idx & 31];
        Sf[idx] += 1.f / (1.f + __expf(-z));
    }
    __syncthreads();

    // ---- motility: MLP(state) -> (64x3) in Zf cols 0..2 ----
    loadX(false); stageW(Wm1, HDIM, SDIM);
    __syncthreads();
    gemm1(1);
    __syncthreads();
    lnrelu(bm1, gm, hm); stageW(Wm2, 3, HDIM);
    __syncthreads();
    gemm2(1);
    __syncthreads();

    // ---- mechanics + integrate + write output (pos | state | conc) ----
    for (int c = tid; c < 64; c += 128) {
        const int i = c0 + c;
        const float am = active[i] ? 1.f : 0.f;
        const float cs = coefsum[i];
#pragma unroll
        for (int dd = 0; dd < 3; ++dd) {
            const float p    = pos[i * 3 + dd];
            const float mech = p * cs - coefp[i * 3 + dd];
            const float mot  = Zf[c * 32 + dd] + bm2[dd];
            out[i * 43 + dd] = p + 0.1f * (mech + mot) * am;
        }
        for (int s = 0; s < SDIM; ++s) out[i * 43 + 3 + s]  = Sf[c * 32 + s];
        for (int m = 0; m < MDIM; ++m) out[i * 43 + 35 + m] = Cc[c * 8 + m];
    }
}

// ---------------------------------------------------------------------------
extern "C" void kernel_launch(void* const* d_in, const int* in_sizes, int n_in,
                              void* d_out, int out_size, void* d_ws, size_t ws_size,
                              hipStream_t stream)
{
    const float* position = (const float*)d_in[0];
    const float* radius   = (const float*)d_in[1];
    const float* state    = (const float*)d_in[2];
    const float* dif      = (const float*)d_in[3];
    const float* deg      = (const float*)d_in[4];
    const float* Wsec1 = (const float*)d_in[5];
    const float* bsec1 = (const float*)d_in[6];
    const float* gsec  = (const float*)d_in[7];
    const float* hsec  = (const float*)d_in[8];
    const float* Wsec2 = (const float*)d_in[9];
    const float* bsec2 = (const float*)d_in[10];
    const float* Wsen1 = (const float*)d_in[11];
    const float* bsen1 = (const float*)d_in[12];
    const float* gsen  = (const float*)d_in[13];
    const float* hsen  = (const float*)d_in[14];
    const float* Wsen2 = (const float*)d_in[15];
    const float* bsen2 = (const float*)d_in[16];
    const float* Wr1 = (const float*)d_in[17];
    const float* br1 = (const float*)d_in[18];
    const float* gr  = (const float*)d_in[19];
    const float* hr  = (const float*)d_in[20];
    const float* Wr2 = (const float*)d_in[21];
    const float* br2 = (const float*)d_in[22];
    const float* Wm1 = (const float*)d_in[23];
    const float* bm1 = (const float*)d_in[24];
    const float* gm  = (const float*)d_in[25];
    const float* hm  = (const float*)d_in[26];
    const float* Wm2 = (const float*)d_in[27];
    const float* bm2 = (const float*)d_in[28];
    const unsigned char* active = (const unsigned char*)d_in[29];

    // workspace layout (floats): sec'(8N) | E(N) | 1/r(N) | conc(8N) | csum(N) | coefp(3N)
    float* ws   = (float*)d_ws;
    float* sec  = ws;
    float* Ebuf = ws + 8 * NC;
    float* invr = ws + 9 * NC;
    float* conc = ws + 10 * NC;
    float* csum = ws + 18 * NC;
    float* cp   = ws + 19 * NC;

    k_zero<<<(12 * NC + 255) / 256, 256, 0, stream>>>(conc, 12 * NC);

    k_secretion<<<NC / 64, 128, 0, stream>>>(state, radius, active, dif,
                                             Wsec1, bsec1, gsec, hsec, Wsec2, bsec2,
                                             sec, Ebuf, invr);

    k_pairwise<<<dim3(NC / 256, 16), 256, 0, stream>>>(position, radius, dif, deg,
                                                       active, sec, Ebuf, invr,
                                                       conc, csum, cp);

    k_update<<<NC / 64, 128, 0, stream>>>(position, state, active,
                                          Wsen1, bsen1, gsen, hsen, Wsen2, bsen2,
                                          Wr1, br1, gr, hr, Wr2, br2,
                                          Wm1, bm1, gm, hm, Wm2, bm2,
                                          conc, csum, cp, (float*)d_out);
}
